// DeepFermi_52175262712182
// MI455X (gfx1250) — matrix operand order repaired and checked by reading, not compiled, not run on hardware
//
#include <hip/hip_runtime.h>
#include <math.h>

typedef __attribute__((ext_vector_type(16))) _Float16 v16h;
typedef __attribute__((ext_vector_type(8)))  _Float16 v8h;
typedef __attribute__((ext_vector_type(8)))  float    v8f;

// ---------------- constants ----------------
#define OSAMP     8
#define MAX_ITER  10
#define NEG_SHIFT 16          // 2*OSAMP
#define OTP       5
#define C_SHARP   500.0f
#define LR        0.1f
#define T_IN      64
#define TOS       512         // OSAMP*T_IN
#define NPIX      16384       // 128*128

// ---------------- workspace layout (float offsets) ----------------
#define WS_CDC       0        // scalar C_dc
#define WS_CNN       1        // 3 floats
#define WS_SPLAM     4
#define WS_TSH       16       // 512
#define WS_STEP      528      // 512
#define WS_CDCPART   1040     // 256 block partials
#define WS_WROW      1296     // 512*64 halves == 16384 floats, [u][j]
#define WS_WCOL      17680    // 64*512 halves == 16384 floats, [j][u]
// total = 34064 floats = ~136 KB

// ---------------- LDS layout (byte offsets), fermi_main ----------------
#define S_WROW   0            // 65536 B  (512*64 f16, [u][j])
#define S_WCOL   65536        // 65536 B  (64*512 f16, [j][u])
#define S_TSH    131072       // 2048 B
#define S_STEP   133120       // 2048 B
#define S_CTCDC  135168       // 64*64 f32 = 16384 B
#define S_ETA    151552       // 3*64 f32 = 768 B
#define S_RESID  152320       // 4 waves * 16*64 f16 = 8192 B
#define S_TOTAL  160512

__device__ __forceinline__ float sigmoidf_(float x) {
    return 1.0f / (1.0f + __expf(-x));
}

// ---- ctc_dc partials: per-pixel baseline-subtract + decimated resize, sum of squares ----
__global__ __launch_bounds__(64) void k_cdc_partial(const float* __restrict__ ctc,
                                                    float* __restrict__ wsf) {
    __shared__ float red[64];
    const int t = threadIdx.x;
    const int p = blockIdx.x * 64 + t;
    const float* c = ctc + (size_t)p * T_IN;
    const float mean = (c[0] + c[1] + c[2] + c[3] + c[4]) * 0.2f;
    // ctc_dc[0] = ctc0[0]; ctc_dc[j] = 0.4375*ctc0[j-1] + 0.5625*ctc0[j]
    float prev = c[0] - mean;
    float acc = prev * prev;
    for (int j = 1; j < T_IN; ++j) {
        float cur = c[j] - mean;
        float v = 0.4375f * prev + 0.5625f * cur;
        acc += v * v;
        prev = cur;
    }
    red[t] = acc;
    __syncthreads();
    for (int s = 32; s > 0; s >>= 1) {
        if (t < s) red[t] += red[t + s];
        __syncthreads();
    }
    if (t == 0) wsf[WS_CDCPART + blockIdx.x] = red[0];
}

// ---- C_nn per channel (deterministic single-block-per-channel reduction) ----
__global__ __launch_bounds__(256) void k_cnn(const float* __restrict__ eta_nn,
                                             float* __restrict__ wsf) {
    __shared__ float red[256];
    const int ch = blockIdx.x;
    const int t = threadIdx.x;
    float acc = 0.f;
    for (int i = t; i < NPIX; i += 256) {
        float v = eta_nn[ch * NPIX + i];
        acc += v * v;
    }
    red[t] = acc;
    __syncthreads();
    for (int s = 128; s > 0; s >>= 1) {
        if (t < s) red[t] += red[t + s];
        __syncthreads();
    }
    if (t == 0) wsf[WS_CNN + ch] = red[0];
}

// ---- scalars, tsh/step tables, W matrix (both layouts, f16) ----
__global__ __launch_bounds__(256) void k_setup(const float* __restrict__ aif,
                                               const float* __restrict__ timev,
                                               const float* __restrict__ lam,
                                               float* __restrict__ wsf) {
    __shared__ float aos[TOS];
    __shared__ float aifm_s;
    const int t = threadIdx.x;
    if (t == 0) {
        float s = 0.f;
        for (int i = 0; i < 256; ++i) s += wsf[WS_CDCPART + i];
        wsf[WS_CDC] = s;
        float x = lam[0];
        wsf[WS_SPLAM] = (x > 20.f) ? x : logf(1.0f + __expf(x));
        aifm_s = (aif[0] + aif[1] + aif[2] + aif[3] + aif[4]) * 0.2f;
    }
    __syncthreads();
    const float aifm = aifm_s;
    // t_os at index NEG_SHIFT: coord (16+0.5)/8-0.5 = 1.5625
    float t16;
    {
        float c = 1.5625f;
        int i0 = (int)floorf(c);
        float w = c - (float)i0;
        int ia = i0 < 0 ? 0 : (i0 > 63 ? 63 : i0);
        int ib = (i0 + 1) < 0 ? 0 : ((i0 + 1) > 63 ? 63 : i0 + 1);
        t16 = (1.f - w) * timev[ia] + w * timev[ib];
    }
    for (int u = t; u < TOS; u += 256) {
        float c = ((float)u + 0.5f) * 0.125f - 0.5f;
        int i0 = (int)floorf(c);
        float w = c - (float)i0;
        int ia = i0 < 0 ? 0 : (i0 > 63 ? 63 : i0);
        int ib = (i0 + 1) < 0 ? 0 : ((i0 + 1) > 63 ? 63 : i0 + 1);
        float tos = (1.f - w) * timev[ia] + w * timev[ib];
        float tsh = tos - t16;
        wsf[WS_TSH + u] = tsh;
        wsf[WS_STEP + u] = sigmoidf_(C_SHARP * tsh);
        aos[u] = (1.f - w) * (aif[ia] - aifm) + w * (aif[ib] - aifm);
    }
    __syncthreads();
    _Float16* wrow = (_Float16*)(wsf + WS_WROW);
    _Float16* wcol = (_Float16*)(wsf + WS_WCOL);
    // W[u][j] = aif_os[16+8j-u]/8 (0 outside)
    for (int idx = t; idx < TOS * 64; idx += 256) {
        int u = idx >> 6;
        int j = idx & 63;
        int tt = NEG_SHIFT + 8 * j - u;
        float v = (tt >= 0 && tt < TOS) ? aos[tt] * 0.125f : 0.0f;
        _Float16 hv = (_Float16)v;
        wrow[u * 64 + j] = hv;
        wcol[j * TOS + u] = hv;
    }
}

// ---- main: 10 GD iterations, 64 pixels per block (16 per wave), WMMA matmuls ----
extern __shared__ char smem[];

__global__ __launch_bounds__(128) void fermi_main(const float* __restrict__ ctc,
                                                  const float* __restrict__ eta_nn,
                                                  const float* __restrict__ wsf,
                                                  float* __restrict__ out) {
    _Float16* wrowL  = (_Float16*)(smem + S_WROW);
    _Float16* wcolL  = (_Float16*)(smem + S_WCOL);
    float*    tshL   = (float*)(smem + S_TSH);
    float*    stepL  = (float*)(smem + S_STEP);
    float*    ctcdcL = (float*)(smem + S_CTCDC);
    float*    etaL   = (float*)(smem + S_ETA);     // [3][64]
    _Float16* residL = (_Float16*)(smem + S_RESID);

    const int tid = threadIdx.x;
    const int lane = tid & 31;
    const int wave = tid >> 5;
    const int pixBase = blockIdx.x * 64;

    // ---- stage W (both layouts, 131072 B contiguous in ws) into LDS via async
    // global->LDS B128 (GVS mode: SGPR base + 32-bit VGPR offset; VDST = LDS
    // byte address; no static LDS in this kernel so dynamic LDS starts at 0).
    {
        const float* gbase = wsf + WS_WROW;
        for (int i = tid; i < 8192; i += 128) {      // 8192 x 16 B
            unsigned goff = (unsigned)(i * 16);
            unsigned laddr = (unsigned)S_WROW + goff;
            asm volatile("global_load_async_to_lds_b128 %0, %1, %2"
                         :: "v"(laddr), "v"(goff), "s"(gbase)
                         : "memory");
        }
        for (int i = tid; i < TOS; i += 128) {
            tshL[i]  = wsf[WS_TSH + i];
            stepL[i] = wsf[WS_STEP + i];
        }
        asm volatile("s_wait_asynccnt 0" ::: "memory");
    }
    // ctc_dc for this block's 64 pixels + eta init
    if (tid < 64) {
        const int p = pixBase + tid;
        const float* c = ctc + (size_t)p * T_IN;
        const float mean = (c[0] + c[1] + c[2] + c[3] + c[4]) * 0.2f;
        float prev = c[0] - mean;
        ctcdcL[tid * 64 + 0] = prev;
        for (int j = 1; j < T_IN; ++j) {
            float cur = c[j] - mean;
            ctcdcL[tid * 64 + j] = 0.4375f * prev + 0.5625f * cur;
            prev = cur;
        }
        for (int ch = 0; ch < 3; ++ch)
            etaL[ch * 64 + tid] = eta_nn[ch * NPIX + p];
    }
    __syncthreads();

    const float invCdc2 = 2.0f / wsf[WS_CDC];
    const float sp_lam  = wsf[WS_SPLAM];
    const float invCnn0 = 1.0f / wsf[WS_CNN + 0];
    const float invCnn1 = 1.0f / wsf[WS_CNN + 1];
    const float invCnn2 = 1.0f / wsf[WS_CNN + 2];

    const int m    = lane & 15;  // M (pixel-in-wave) for A/B fragments, N for C tiles
    const int half = lane >> 4;
    const int pixw = wave * 16;  // wave's pixel offset within block

    // writer lanes: (lane&15)<8 -> pixel wp = (lane&7) + 8*half
    const int  wr = lane & 7;
    const bool writer = (m < 8);
    const int  wp = wr + 8 * half;
    const int  wgp = pixBase + pixw + wp;
    const float prior0 = eta_nn[0 * NPIX + wgp];
    const float prior1 = eta_nn[1 * NPIX + wgp];
    const float prior2 = eta_nn[2 * NPIX + wgp];

    _Float16* resW = residL + wave * (16 * 64);

    union F8 { float4 q[2]; float f[8]; };

    for (int it = 0; it < MAX_ITER; ++it) {
        // eta for A-fragment row m, and for the 8 C-layout rows of this half
        const float myA  = etaL[0 * 64 + pixw + m];
        const float myK  = etaL[1 * 64 + pixw + m];
        const float myT0 = etaL[2 * 64 + pixw + m];
        float eAr[8], eKr[8], eTr[8];
#pragma unroll
        for (int r = 0; r < 8; ++r) {
            int p = pixw + r + 8 * half;
            eAr[r] = etaL[0 * 64 + p];
            eKr[r] = etaL[1 * 64 + p];
            eTr[r] = etaL[2 * 64 + p];
        }

        // ---------- forward: ctc_est = IR(16x512) * W(512x64) ----------
        v8f cacc[4] = {};
        for (int kc = 0; kc < 16; ++kc) {
            // A-fragment u-mapping: e<8 -> u = ubase+e; e>=8 -> u = ubase+16+(e-8)
            const int ubase = kc * 32 + 8 * half;
            F8 tlo, thi, slo, shi;
            tlo.q[0] = *(const float4*)(tshL + ubase);
            tlo.q[1] = *(const float4*)(tshL + ubase + 4);
            thi.q[0] = *(const float4*)(tshL + ubase + 16);
            thi.q[1] = *(const float4*)(tshL + ubase + 20);
            slo.q[0] = *(const float4*)(stepL + ubase);
            slo.q[1] = *(const float4*)(stepL + ubase + 4);
            shi.q[0] = *(const float4*)(stepL + ubase + 16);
            shi.q[1] = *(const float4*)(stepL + ubase + 20);

            union { v16h v; _Float16 h[16]; } af;
#pragma unroll
            for (int e = 0; e < 8; ++e) {
                float s = sigmoidf_(myK * (myT0 - tlo.f[e]));
                af.h[e] = (_Float16)(myA * s * slo.f[e]);
            }
#pragma unroll
            for (int e = 0; e < 8; ++e) {
                float s = sigmoidf_(myK * (myT0 - thi.f[e]));
                af.h[8 + e] = (_Float16)(myA * s * shi.f[e]);
            }
            const int u0 = kc * 32 + 8 * half;
#pragma unroll
            for (int nt = 0; nt < 4; ++nt) {
                union { v16h v; v8h p2[2]; } bb;
                bb.p2[0] = *(const v8h*)(wcolL + (nt * 16 + m) * TOS + u0);
                bb.p2[1] = *(const v8h*)(wcolL + (nt * 16 + m) * TOS + u0 + 16);
                v16h b = bb.v;
                cacc[nt] = __builtin_amdgcn_wmma_f32_16x16x32_f16(
                    false, af.v, false, b, (short)0, cacc[nt], false, false);
            }
        }
        // residual (C layout -> plain [p][j] f16 in LDS)
#pragma unroll
        for (int nt = 0; nt < 4; ++nt) {
            const int j = nt * 16 + m;
#pragma unroll
            for (int r = 0; r < 8; ++r) {
                int p = r + 8 * half;
                float rr = cacc[nt][r] - ctcdcL[(pixw + p) * 64 + j];
                resW[p * 64 + j] = (_Float16)rr;
            }
        }
        __builtin_amdgcn_wave_barrier();  // same-wave LDS ordering is in-order; keep scheduler honest

        // ---------- backward: d_ir = resid(16x64) * W^T(64x512), fused chain ----------
        float accA[8] = {}, accK[8] = {}, accT[8] = {};
        for (int ut = 0; ut < 32; ++ut) {
            v8f d = {};
#pragma unroll
            for (int kc2 = 0; kc2 < 2; ++kc2) {
                // A-fragment of resid: element e -> j = kc2*32 + ((e<8?e:e+8) + 8*half)
                const int jlo = kc2 * 32 + 8 * half;
                union { v16h v; v8h p2[2]; } a2;
                a2.p2[0] = *(const v8h*)(resW + m * 64 + jlo);
                a2.p2[1] = *(const v8h*)(resW + m * 64 + jlo + 16);
                // B-fragment: B[k=j][n=u] = W[u][j]; element e -> j = kc2*32 + ((e<8?e:e+8) + 8*half)
                const int u = ut * 16 + m;
                union { v16h v; v8h p2[2]; } bb2;
                bb2.p2[0] = *(const v8h*)(wrowL + u * 64 + jlo);
                bb2.p2[1] = *(const v8h*)(wrowL + u * 64 + jlo + 16);
                v16h b2 = bb2.v;
                d = __builtin_amdgcn_wmma_f32_16x16x32_f16(
                    false, a2.v, false, b2, (short)0, d, false, false);
            }
            const int u = ut * 16 + m;
            const float tshu = tshL[u];
            const float stepu = stepL[u];
#pragma unroll
            for (int r = 0; r < 8; ++r) {
                float g  = d[r] * invCdc2;       // dL/d ir[p,u]
                float dt = eTr[r] - tshu;
                float s  = sigmoidf_(eKr[r] * dt);
                float gs = g * stepu;
                accA[r] += gs * s;
                float t1 = gs * eAr[r] * s * (1.0f - s);
                accK[r] += t1 * dt;
                accT[r] += t1 * eKr[r];
            }
        }
        // butterfly sum over the 16 lanes of each half (u-partials)
#pragma unroll
        for (int mask = 1; mask < 16; mask <<= 1) {
#pragma unroll
            for (int r = 0; r < 8; ++r) {
                accA[r] += __shfl_xor(accA[r], mask, 32);
                accK[r] += __shfl_xor(accK[r], mask, 32);
                accT[r] += __shfl_xor(accT[r], mask, 32);
            }
        }
        // eta update (writer lanes: 0..7 and 16..23)
        if (writer) {
            float e0 = etaL[0 * 64 + pixw + wp];
            float e1 = etaL[1 * 64 + pixw + wp];
            float e2 = etaL[2 * 64 + pixw + wp];
            float g0 = accA[wr] + 2.0f * sp_lam * (e0 - prior0) * invCnn0 + ((e0 < 0.f) ? 2.0f * e0 : 0.f);
            float g1 = accK[wr] + 2.0f * sp_lam * (e1 - prior1) * invCnn1 + ((e1 < 0.f) ? 2.0f * e1 : 0.f);
            float g2 = accT[wr] + 2.0f * sp_lam * (e2 - prior2) * invCnn2 + ((e2 < 0.f) ? 2.0f * e2 : 0.f);
            etaL[0 * 64 + pixw + wp] = e0 - LR * g0;
            etaL[1 * 64 + pixw + wp] = e1 - LR * g1;
            etaL[2 * 64 + pixw + wp] = e2 - LR * g2;
        }
        __syncthreads();
    }

    if (tid < 64) {
        const int p = pixBase + tid;
        out[0 * NPIX + p] = etaL[0 * 64 + tid];
        out[1 * NPIX + p] = etaL[1 * 64 + tid];
        out[2 * NPIX + p] = etaL[2 * 64 + tid];
    }
}

extern "C" void kernel_launch(void* const* d_in, const int* in_sizes, int n_in,
                              void* d_out, int out_size, void* d_ws, size_t ws_size,
                              hipStream_t stream) {
    (void)in_sizes; (void)n_in; (void)out_size; (void)ws_size;
    const float* ctc   = (const float*)d_in[0];   // (1,128,128,64)
    const float* aif   = (const float*)d_in[1];   // (1,1,1,64)
    const float* timev = (const float*)d_in[2];   // (64,)
    // d_in[3] = seg (unused)
    const float* eta   = (const float*)d_in[4];   // (1,3,128,128)
    const float* lam   = (const float*)d_in[5];   // (1,)
    float* out = (float*)d_out;
    float* wsf = (float*)d_ws;

    k_cdc_partial<<<256, 64, 0, stream>>>(ctc, wsf);
    k_cnn<<<3, 256, 0, stream>>>(eta, wsf);
    k_setup<<<1, 256, 0, stream>>>(aif, timev, lam, wsf);
    fermi_main<<<256, 128, S_TOTAL, stream>>>(ctc, eta, wsf, out);
}
